// QuantizedConv2d_42013370090057
// MI455X (gfx1250) — 8-bit matrix operand order repaired and checked by reading, compiled, not run on hardware
//
#include <hip/hip_runtime.h>

typedef int v8i __attribute__((ext_vector_type(8)));

#define QC_IN_ZP_SPLAT 0x03030303u
#define QC_OUT_ZP      -2.0f
#define QC_SCALE_RATIO 0.5f

__global__ void pack_x_nhwc(const int* __restrict__ src,
                            signed char* __restrict__ dst) {
  const int t   = blockIdx.x * blockDim.x + threadIdx.x;
  const int c   = (t & 31) * 4;
  const int pix = t >> 5;
  const int n   = pix / 3136;
  const int hw  = pix - n * 3136;
  const int* s  = src + ((size_t)n * 128 + c) * 3136 + hw;
  unsigned int v = ((unsigned int)s[0]        & 0xFFu)
                 | (((unsigned int)s[3136]    & 0xFFu) << 8)
                 | (((unsigned int)s[2*3136]  & 0xFFu) << 16)
                 | (((unsigned int)s[3*3136]  & 0xFFu) << 24);
  *(unsigned int*)(dst + (size_t)pix * 128 + c) = v;
}

__global__ void pack_w_step(const int* __restrict__ src,
                            signed char* __restrict__ dst) {
  const int t    = blockIdx.x * blockDim.x + threadIdx.x;
  const int ci   = (t & 31) * 4;
  const int rest = t >> 5;
  const int tap  = rest % 9;
  const int co   = rest / 9;
  const int* s   = src + (size_t)co * 1152 + ci * 9 + tap;
  unsigned int v = ((unsigned int)s[0]  & 0xFFu)
                 | (((unsigned int)s[9]  & 0xFFu) << 8)
                 | (((unsigned int)s[18] & 0xFFu) << 16)
                 | (((unsigned int)s[27] & 0xFFu) << 24);
  const int step = tap * 2 + (ci >> 6);
  *(unsigned int*)(dst + ((size_t)step * 256 + co) * 64 + (ci & 63)) = v;
}

__global__ __launch_bounds__(256) void conv3x3_s8_wmma(
    const signed char* __restrict__ xq,
    const signed char* __restrict__ wq,
    const int*         __restrict__ bias,
    const float*       __restrict__ wscale,
    float*             __restrict__ out)
{
  constexpr int HH = 56, WW = 56, COUT = 256, NSTEP = 18;
  constexpr int HWP = HH * WW;

  __shared__ alignas(16) signed char sA[2][128 * 64];
  __shared__ alignas(16) signed char sB[2][256 * 64];

  const int tid  = threadIdx.x;
  const int lane = tid & 31;
  const int wave = tid >> 5;
  const int mw   = wave & 3;
  const int nw   = wave >> 2;

  const int m_base = blockIdx.x * 128;

  const int ar    = tid >> 1;
  const int ah    = (tid & 1) * 32;
  const int p     = m_base + ar;
  const int n_img = p / HWP;
  const int hw    = p - n_img * HWP;
  const int oh    = hw / WW;
  const int ow    = hw - oh * WW;

  const signed char* wrow = wq + (size_t)tid * 64;

  v8i acc[2][8] = {};

  {
    uint4 aR[2], bR[4];
    const int ih = oh - 1, iw = ow - 1;
    if ((unsigned)ih < (unsigned)HH && (unsigned)iw < (unsigned)WW) {
      const uint4* g =
          (const uint4*)(xq + (((size_t)n_img * HH + ih) * WW + iw) * 128 + ah);
      aR[0] = g[0]; aR[1] = g[1];
    } else {
      aR[0] = aR[1] = make_uint4(QC_IN_ZP_SPLAT, QC_IN_ZP_SPLAT,
                                 QC_IN_ZP_SPLAT, QC_IN_ZP_SPLAT);
    }
    const uint4* g = (const uint4*)(wrow);
    bR[0] = g[0]; bR[1] = g[1]; bR[2] = g[2]; bR[3] = g[3];
    *(uint4*)(&sA[0][ar * 64 + ah])      = aR[0];
    *(uint4*)(&sA[0][ar * 64 + ah + 16]) = aR[1];
#pragma unroll
    for (int q = 0; q < 4; ++q)
      *(uint4*)(&sB[0][tid * 64 + q * 16]) = bR[q];
  }

  for (int s = 0; s < NSTEP; ++s) {
    __syncthreads();
    const int buf = s & 1;

    uint4 aN[2], bN[4];
    const bool more = (s + 1) < NSTEP;
    if (more) {
      const int s1  = s + 1;
      const int tap = s1 >> 1;
      const int kh  = tap / 3;
      const int kw  = tap - kh * 3;
      const int cib = (s1 & 1) * 64 + ah;
      const int ih  = oh + kh - 1;
      const int iw  = ow + kw - 1;
      if ((unsigned)ih < (unsigned)HH && (unsigned)iw < (unsigned)WW) {
        const uint4* g =
            (const uint4*)(xq + (((size_t)n_img * HH + ih) * WW + iw) * 128 + cib);
        aN[0] = g[0]; aN[1] = g[1];
      } else {
        aN[0] = aN[1] = make_uint4(QC_IN_ZP_SPLAT, QC_IN_ZP_SPLAT,
                                   QC_IN_ZP_SPLAT, QC_IN_ZP_SPLAT);
      }
      const uint4* g = (const uint4*)(wrow + (size_t)s1 * 16384);
      bN[0] = g[0]; bN[1] = g[1]; bN[2] = g[2]; bN[3] = g[3];
    }

    v8i afrag[2];
#pragma unroll
    for (int i = 0; i < 2; ++i) {
      const int row = mw * 32 + i * 16 + (lane & 15);
      const unsigned int* a32 =
          (const unsigned int*)(&sA[buf][row * 64 + ((lane & 16) ? 16 : 0)]);
      afrag[i][0] = a32[0];  afrag[i][1] = a32[1];
      afrag[i][2] = a32[2];  afrag[i][3] = a32[3];
      afrag[i][4] = a32[8];  afrag[i][5] = a32[9];
      afrag[i][6] = a32[10]; afrag[i][7] = a32[11];
    }

    v8i bfr[2];
    {
      const int col = nw * 128 + (lane & 15);
      const unsigned int* b32 =
          (const unsigned int*)(&sB[buf][col * 64 + ((lane & 16) ? 16 : 0)]);
      bfr[0][0] = b32[0]; bfr[0][1] = b32[1]; bfr[0][2] = b32[2];  bfr[0][3] = b32[3];
      bfr[0][4] = b32[8]; bfr[0][5] = b32[9]; bfr[0][6] = b32[10]; bfr[0][7] = b32[11];
    }
#pragma unroll
    for (int f = 0; f < 8; ++f) {
      if (f + 1 < 8) {
        const int col = nw * 128 + (f + 1) * 16 + (lane & 15);
        const unsigned int* b32 =
            (const unsigned int*)(&sB[buf][col * 64 + ((lane & 16) ? 16 : 0)]);
        v8i& d = bfr[(f + 1) & 1];
        d[0] = b32[0]; d[1] = b32[1]; d[2] = b32[2];  d[3] = b32[3];
        d[4] = b32[8]; d[5] = b32[9]; d[6] = b32[10]; d[7] = b32[11];
      }
#pragma unroll
      for (int i = 0; i < 2; ++i) {
        acc[i][f] = __builtin_amdgcn_wmma_i32_16x16x64_iu8(
             true, afrag[i],  true, bfr[f & 1],
            acc[i][f],  false,  false);
      }
    }

    if (more) {
      const int nbuf = (s + 1) & 1;
      *(uint4*)(&sA[nbuf][ar * 64 + ah])      = aN[0];
      *(uint4*)(&sA[nbuf][ar * 64 + ah + 16]) = aN[1];
#pragma unroll
      for (int q = 0; q < 4; ++q)
        *(uint4*)(&sB[nbuf][tid * 64 + q * 16]) = bN[q];
    }
  }

  int   colv[8];
  float sv[8];
  int   bv[8];
#pragma unroll
  for (int f = 0; f < 8; ++f) {
    colv[f] = nw * 128 + f * 16 + (lane & 15);
    sv[f]   = QC_SCALE_RATIO * wscale[colv[f]];
    bv[f]   = bias[colv[f]];
  }
#pragma unroll
  for (int i = 0; i < 2; ++i) {
#pragma unroll
    for (int v = 0; v < 8; ++v) {
      const int row = m_base + mw * 32 + i * 16 + v + ((lane & 16) ? 8 : 0);
      const int n2  = row / HWP;
      const int hw2 = row - n2 * HWP;
      const size_t obase = (size_t)n2 * COUT * HWP + hw2;
#pragma unroll
      for (int f = 0; f < 8; ++f) {
        float af = (float)(acc[i][f][v] + bv[f]);
        float y  = __fadd_rn(__fmul_rn(af, sv[f]), QC_OUT_ZP);
        y = rintf(y);
        y = fminf(fmaxf(y, -128.0f), 127.0f);
        out[obase + (size_t)colv[f] * HWP] = y;
      }
    }
  }
}

extern "C" void kernel_launch(void* const* d_in, const int* in_sizes, int n_in,
                              void* d_out, int out_size, void* d_ws, size_t ws_size,
                              hipStream_t stream) {
  const int*   x      = (const int*)d_in[0];
  const int*   w      = (const int*)d_in[1];
  const int*   bias   = (const int*)d_in[2];
  const float* wscale = (const float*)d_in[3];
  float*       out    = (float*)d_out;

  constexpr int W_ELEMS = 256 * 128 * 3 * 3;
  constexpr int X_ELEMS = 32 * 128 * 56 * 56;

  signed char* wq = (signed char*)d_ws;
  signed char* xq = (signed char*)d_ws + W_ELEMS;

  pack_w_step<<<(W_ELEMS / 4) / 256, 256, 0, stream>>>(w, wq);
  pack_x_nhwc<<<(X_ELEMS / 4) / 256, 256, 0, stream>>>(x, xq);

  conv3x3_s8_wmma<<<784, 256, 0, stream>>>(xq, wq, bias, wscale, out);
}
